// LegacyComplexVarGAT_40690520163149
// MI455X (gfx1250) — matrix operand order repaired and checked by reading, not compiled, not run on hardware
//
#include <hip/hip_runtime.h>
#include <hip/hip_bf16.h>

// ---------------- types for WMMA ----------------
typedef __attribute__((ext_vector_type(16))) __bf16 v16bf;
typedef __attribute__((ext_vector_type(8)))  float  v8f;

#define HID 128
#define HEADS 4
#define HD 32
#define MT_PER_BLOCK 8   // 16-row tiles per block (B stays in registers; A double-buffered)

static __device__ __forceinline__ unsigned short f2bf(float f) {
    unsigned u = __float_as_uint(f);
    unsigned r = u + 0x7FFFu + ((u >> 16) & 1u);   // round-to-nearest-even
    return (unsigned short)(r >> 16);
}
static __device__ __forceinline__ unsigned f2ord(float f) {
    unsigned u = __float_as_uint(f);
    return (u & 0x80000000u) ? ~u : (u | 0x80000000u);
}
static __device__ __forceinline__ float ord2f(unsigned u) {
    unsigned v = (u & 0x80000000u) ? (u & 0x7FFFFFFFu) : ~u;
    return __uint_as_float(v);
}

// ---------------- fills / converts ----------------
__global__ void k_fill_f32(float* p, size_t n, float v) {
    size_t i = (size_t)blockIdx.x * blockDim.x + threadIdx.x;
    if (i < n) p[i] = v;
}
__global__ void k_fill_u32(unsigned* p, size_t n, unsigned v) {
    size_t i = (size_t)blockIdx.x * blockDim.x + threadIdx.x;
    if (i < n) p[i] = v;
}
__global__ void k_f32_to_bf16(const float* __restrict__ in, unsigned short* __restrict__ out, size_t n) {
    size_t i = (size_t)blockIdx.x * blockDim.x + threadIdx.x;
    if (i < n) out[i] = f2bf(in[i]);
}

// Pack f32 weight matrices [m][128][128] (row-major K x N) into per-lane WMMA-B layout:
//   Bpk[m][kk][tile][lane][j]  (j=0..15 contiguous bf16 for one lane)
// where k = kk*32 + (lane>>4)*16 + j, col = tile*16 + (lane&15).
__global__ void k_pack_w(const float* __restrict__ W, unsigned short* __restrict__ Bpk, int nmat) {
    int i = blockIdx.x * blockDim.x + threadIdx.x;
    if (i >= nmat * HID * HID) return;
    int m    = i >> 14;           // matrix index
    int r    = i & 16383;
    int j    = r & 15;
    int lane = (r >> 4) & 31;
    int t    = (r >> 9) & 7;
    int kk   = (r >> 12) & 3;
    int k    = kk * 32 + (lane >> 4) * 16 + j;
    int col  = t * 16 + (lane & 15);
    Bpk[i] = f2bf(W[(size_t)m * HID * HID + k * HID + col]);
}

// ---------------- WMMA GEMM: C[M,128] = A_bf16[M,128] @ W[128,128] (+bias) ----------------
// Block = 256 threads = 8 waves; wave w owns column tile w. B (all 4 K-chunks) loaded once
// into registers. A is software-pipelined with two register tile buffers: tile t+1's
// 8 x b128 loads are issued while tile t's 4 x v_wmma_f32_16x16x32_bf16 execute, so the
// compiler can use partial s_wait_loadcnt instead of a full drain before every WMMA.
__global__ void k_gemm_bf16_k128(const unsigned short* __restrict__ A,
                                 const unsigned short* __restrict__ Bpk,
                                 const float* __restrict__ bias,
                                 float* __restrict__ C, int M) {
    const int wv   = threadIdx.x >> 5;
    const int lane = threadIdx.x & 31;
    const int lm   = lane & 15;
    const int hi   = lane >> 4;

    union V16 { v16bf v; unsigned short u[16]; uint4 q[2]; };

    // load all of B for this column tile into registers (packed layout -> contiguous)
    V16 b[4];
    const unsigned short* bp = Bpk + ((size_t)wv * 32 + lane) * 16;
#pragma unroll
    for (int kk = 0; kk < 4; ++kk) {
        b[kk].q[0] = *(const uint4*)(bp + kk * 4096);
        b[kk].q[1] = *(const uint4*)(bp + kk * 4096 + 8);
    }
    const int col = wv * 16 + lm;
    const float bv = bias ? bias[col] : 0.f;

    const int tileBase = blockIdx.x * MT_PER_BLOCK;
    const int rb0 = tileBase * 16;
    if (rb0 >= M) return;

    V16 abuf[2][4];
    {
        const unsigned short* arow = A + (size_t)(rb0 + lm) * HID;
#pragma unroll
        for (int kk = 0; kk < 4; ++kk) {
            const int koff = kk * 32 + hi * 16;
            abuf[0][kk].q[0] = *(const uint4*)(arow + koff);
            abuf[0][kk].q[1] = *(const uint4*)(arow + koff + 8);
        }
    }

#pragma unroll
    for (int t = 0; t < MT_PER_BLOCK; ++t) {
        const int rowBase = (tileBase + t) * 16;
        if (rowBase >= M) return;
        // prefetch next tile's A into the other buffer
        const int nrb = rowBase + 16;
        if (t + 1 < MT_PER_BLOCK && nrb < M) {
            const unsigned short* arow = A + (size_t)(nrb + lm) * HID;
#pragma unroll
            for (int kk = 0; kk < 4; ++kk) {
                const int koff = kk * 32 + hi * 16;
                abuf[(t + 1) & 1][kk].q[0] = *(const uint4*)(arow + koff);
                abuf[(t + 1) & 1][kk].q[1] = *(const uint4*)(arow + koff + 8);
            }
        }
        v8f c;
#pragma unroll
        for (int r = 0; r < 8; ++r) c[r] = bv;
#pragma unroll
        for (int kk = 0; kk < 4; ++kk) {
            c = __builtin_amdgcn_wmma_f32_16x16x32_bf16(false, abuf[t & 1][kk].v, false, b[kk].v,
                                                        (short)0, c, false, false);
        }
#pragma unroll
        for (int r = 0; r < 8; ++r) {
            int row = rowBase + r + hi * 8;
            C[(size_t)row * HID + col] = c[r];
        }
    }
}

// ---------------- projections ----------------
__global__ void k_node_proj(const float* __restrict__ x, const float* __restrict__ Wp,
                            const float* __restrict__ bp, float* __restrict__ h, int N) {
    int i = blockIdx.x * blockDim.x + threadIdx.x;
    if (i >= N * HID) return;
    int n = i >> 7, f = i & 127;
    float acc = bp[f];
    const float* xr = x + (size_t)n * 28;
#pragma unroll
    for (int k = 0; k < 28; ++k) acc += xr[k] * Wp[k * HID + f];
    h[i] = acc;
}
__global__ void k_edge_proj(const float* __restrict__ ea, const float* __restrict__ Wp,
                            const float* __restrict__ bp, float* __restrict__ ee, int E) {
    int i = blockIdx.x * blockDim.x + threadIdx.x;
    if (i >= E * HID) return;
    int e = i >> 7, f = i & 127;
    const float* a = ea + (size_t)e * 3;
    float acc = bp[f] + a[0] * Wp[0 * HID + f] + a[1] * Wp[1 * HID + f] + a[2] * Wp[2 * HID + f];
    ee[(size_t)e * HID + f] = acc;
}

// ---------------- self-loop edge features (PyG fill_value='mean') ----------------
__global__ void k_selfloop_accum(const float* __restrict__ ee, const int* __restrict__ ei,
                                 float* __restrict__ loopsum, float* __restrict__ cnt, int E) {
    int i = blockIdx.x * blockDim.x + threadIdx.x;
    if (i >= E * HID) return;
    int e = i >> 7, f = i & 127;
    int d = ei[E + e];
    atomicAdd(&loopsum[(size_t)d * HID + f], ee[(size_t)e * HID + f]);
    if (f == 0) atomicAdd(&cnt[d], 1.0f);
}
__global__ void k_selfloop_div(float* __restrict__ loopsum, const float* __restrict__ cnt, int N) {
    int i = blockIdx.x * blockDim.x + threadIdx.x;
    if (i >= N * HID) return;
    int n = i >> 7;
    loopsum[i] /= fmaxf(cnt[n], 1.0f);
}

// ---------------- GAT edge passes ----------------
__global__ void k_edge_logits(const float* __restrict__ xl, const float* __restrict__ xr,
                              const float* __restrict__ em, const int* __restrict__ ei,
                              const float* __restrict__ att, float* __restrict__ logits,
                              unsigned* __restrict__ emax, int E, int N) {
    int i = blockIdx.x * blockDim.x + threadIdx.x;
    int Et = E + N;
    if (i >= Et * HEADS) return;
    int e = i >> 2, hh = i & 3;
    int s, d;
    if (e < E) { s = ei[e]; d = ei[E + e]; } else { s = d = e - E; }
    const float4* a  = (const float4*)(att + hh * HD);
    const float4* pl = (const float4*)(xl + (size_t)s * HID + hh * HD);
    const float4* pr = (const float4*)(xr + (size_t)d * HID + hh * HD);
    const float4* pe = (const float4*)(em + (size_t)e * HID + hh * HD);
    float acc = 0.f;
#pragma unroll
    for (int q = 0; q < 8; ++q) {
        float4 va = a[q], vl = pl[q], vr = pr[q], ve = pe[q];
        float m0 = vl.x + vr.x + ve.x; m0 = m0 > 0.f ? m0 : 0.2f * m0;
        float m1 = vl.y + vr.y + ve.y; m1 = m1 > 0.f ? m1 : 0.2f * m1;
        float m2 = vl.z + vr.z + ve.z; m2 = m2 > 0.f ? m2 : 0.2f * m2;
        float m3 = vl.w + vr.w + ve.w; m3 = m3 > 0.f ? m3 : 0.2f * m3;
        acc += m0 * va.x + m1 * va.y + m2 * va.z + m3 * va.w;
    }
    logits[i] = acc;
    atomicMax(&emax[d * HEADS + hh], f2ord(acc));
}
// one thread per (edge, feature): coalesced xl loads and gout atomics
__global__ void k_edge_accum(const float* __restrict__ xl, const int* __restrict__ ei,
                             const float* __restrict__ logits, const unsigned* __restrict__ emax,
                             float* __restrict__ denom, float* __restrict__ gout, int E, int N) {
    int i = blockIdx.x * blockDim.x + threadIdx.x;
    int Et = E + N;
    if (i >= Et * HID) return;
    int e = i >> 7, f = i & 127;
    int hh = f >> 5;
    int s, d;
    if (e < E) { s = ei[e]; d = ei[E + e]; } else { s = d = e - E; }
    float w = __expf(logits[e * HEADS + hh] - ord2f(emax[d * HEADS + hh]));
    if ((f & 31) == 0) atomicAdd(&denom[d * HEADS + hh], w);
    atomicAdd(&gout[(size_t)d * HID + f], w * xl[(size_t)s * HID + f]);
}
__global__ void k_gat_finalize(float* __restrict__ gout, const float* __restrict__ denom,
                               const float* __restrict__ gbias, int N) {
    int i = blockIdx.x * blockDim.x + threadIdx.x;
    if (i >= N * HID) return;
    int n = i >> 7, f = i & 127;
    gout[i] = gout[i] / (denom[n * HEADS + (f >> 5)] + 1e-16f) + gbias[f];
}

// ---------------- batchnorm over N nodes ----------------
__global__ void k_bn_stats(const float* __restrict__ g, float* __restrict__ mean,
                           float* __restrict__ var, int N) {
    int f = blockIdx.x;                 // one block per feature (128 blocks)
    __shared__ float ssum[256], ssq[256];
    float s = 0.f, q = 0.f;
    for (int n = threadIdx.x; n < N; n += blockDim.x) {
        float v = g[(size_t)n * HID + f];
        s += v; q += v * v;
    }
    ssum[threadIdx.x] = s; ssq[threadIdx.x] = q;
    __syncthreads();
    for (int st = 128; st > 0; st >>= 1) {
        if ((int)threadIdx.x < st) {
            ssum[threadIdx.x] += ssum[threadIdx.x + st];
            ssq[threadIdx.x]  += ssq[threadIdx.x + st];
        }
        __syncthreads();
    }
    if (threadIdx.x == 0) {
        float m = ssum[0] / N;
        mean[f] = m;
        var[f]  = ssq[0] / N - m * m;   // biased, matches torch BN
    }
}
__global__ void k_bn_apply_relu_res(const float* __restrict__ g, const float* __restrict__ mean,
                                    const float* __restrict__ var, const float* __restrict__ gam,
                                    const float* __restrict__ bet, const float* __restrict__ hold,
                                    float* __restrict__ hnew, int N) {
    int i = blockIdx.x * blockDim.x + threadIdx.x;
    if (i >= N * HID) return;
    int f = i & 127;
    float v = (g[i] - mean[f]) * gam[f] * rsqrtf(var[f] + 1e-5f) + bet[f];
    hnew[i] = fmaxf(v, 0.f) + hold[i];
}

// ---------------- MLP head (tiny: B=64) ----------------
__global__ void k_head_lin1(const float* __restrict__ h, const int* __restrict__ mut,
                            const float* __restrict__ pert, const float* __restrict__ W,
                            const float* __restrict__ b, float* __restrict__ out) {
    int i = blockIdx.x * blockDim.x + threadIdx.x;     // [64,256]
    if (i >= 64 * 256) return;
    int bb = i >> 8, j = i & 255;
    const float* hr = h + (size_t)mut[bb] * HID;
    float acc = b[j];
    for (int k = 0; k < HID; ++k) acc += hr[k] * W[k * 256 + j];
    const float* pr = pert + bb * 9;
    for (int k = 0; k < 9; ++k)  acc += pr[k] * W[(HID + k) * 256 + j];
    out[i] = acc;
}
__global__ void k_head_bn_relu(float* __restrict__ t, const float* __restrict__ gam,
                               const float* __restrict__ bet, int B, int F) {
    int j = threadIdx.x;
    if (j >= F) return;
    float s = 0.f, q = 0.f;
    for (int bb = 0; bb < B; ++bb) { float v = t[bb * F + j]; s += v; q += v * v; }
    float m = s / B, vv = q / B - m * m;
    float sc = gam[j] * rsqrtf(vv + 1e-5f);
    for (int bb = 0; bb < B; ++bb) {
        float v = (t[bb * F + j] - m) * sc + bet[j];
        t[bb * F + j] = fmaxf(v, 0.f);
    }
}
__global__ void k_head_lin(const float* __restrict__ in, const float* __restrict__ W,
                           const float* __restrict__ bias, float* __restrict__ out,
                           int B, int K, int J, int relu) {
    int i = blockIdx.x * blockDim.x + threadIdx.x;
    if (i >= B * J) return;
    int bb = i / J, j = i % J;
    float acc = bias[j];
    for (int k = 0; k < K; ++k) acc += in[bb * K + k] * W[k * J + j];
    if (relu) acc = fmaxf(acc, 0.f);
    out[i] = acc;
}
__global__ void k_head_out(const float* __restrict__ t3, const float* __restrict__ clsW,
                           const float* __restrict__ clsb, const float* __restrict__ regW,
                           const float* __restrict__ regb, float* __restrict__ out) {
    int bb = threadIdx.x;
    if (bb >= 64) return;
    const float* r = t3 + bb * 64;
    float c = clsb[0], g = regb[0];
    for (int k = 0; k < 64; ++k) { c += r[k] * clsW[k]; g += r[k] * regW[k]; }
    out[bb * 2 + 0] = c;
    out[bb * 2 + 1] = g;
}

// ---------------- launcher ----------------
extern "C" void kernel_launch(void* const* d_in, const int* in_sizes, int n_in,
                              void* d_out, int out_size, void* d_ws, size_t ws_size,
                              hipStream_t stream) {
    const int N = in_sizes[0] / 28;       // 30000
    const int E = in_sizes[2] / 2;        // 960000
    const int Et = E + N;
    const int L = 3, B = 64;

    const float* x    = (const float*)d_in[0];
    const float* ea   = (const float*)d_in[1];
    const int*   ei   = (const int*)  d_in[2];
    const int*   mut  = (const int*)  d_in[3];
    const float* pert = (const float*)d_in[4];
    const float* npW  = (const float*)d_in[5];
    const float* npB  = (const float*)d_in[6];
    const float* epW  = (const float*)d_in[7];
    const float* epB  = (const float*)d_in[8];
    const float* Wl   = (const float*)d_in[9];
    const float* bl   = (const float*)d_in[10];
    const float* Wr   = (const float*)d_in[11];
    const float* br   = (const float*)d_in[12];
    const float* We   = (const float*)d_in[13];
    const float* att  = (const float*)d_in[14];
    const float* gbias= (const float*)d_in[15];
    const float* bng  = (const float*)d_in[16];
    const float* bnb  = (const float*)d_in[17];
    const float* r1W  = (const float*)d_in[18];
    const float* r1b  = (const float*)d_in[19];
    const float* bn1g = (const float*)d_in[20];
    const float* bn1b = (const float*)d_in[21];
    const float* r2W  = (const float*)d_in[22];
    const float* r2b  = (const float*)d_in[23];
    const float* bn2g = (const float*)d_in[24];
    const float* bn2b = (const float*)d_in[25];
    const float* r3W  = (const float*)d_in[26];
    const float* r3b  = (const float*)d_in[27];
    const float* clsW = (const float*)d_in[28];
    const float* clsb = (const float*)d_in[29];
    const float* regW = (const float*)d_in[30];
    const float* regb = (const float*)d_in[31];
    float* out = (float*)d_out;

    // bump allocator over d_ws
    char* base = (char*)d_ws;
    size_t off = 0;
    auto alloc = [&](size_t bytes) -> void* {
        off = (off + 255) & ~(size_t)255;
        void* p = base + off;
        off += bytes;
        return p;
    };
    float* ee     = (float*)alloc((size_t)Et * HID * 4);   // rows [E,Et) = loop_attr
    float* em     = (float*)alloc((size_t)Et * HID * 4);
    unsigned short* eebf = (unsigned short*)alloc((size_t)Et * HID * 2);
    float* hA     = (float*)alloc((size_t)N * HID * 4);
    float* hB     = (float*)alloc((size_t)N * HID * 4);
    unsigned short* hbf  = (unsigned short*)alloc((size_t)N * HID * 2);
    float* xl     = (float*)alloc((size_t)N * HID * 4);
    float* xr     = (float*)alloc((size_t)N * HID * 4);
    float* gout   = (float*)alloc((size_t)N * HID * 4);
    float* logits = (float*)alloc((size_t)Et * HEADS * 4);
    unsigned* emax= (unsigned*)alloc((size_t)N * HEADS * 4);
    float* denom  = (float*)alloc((size_t)N * HEADS * 4);
    float* cnt    = (float*)alloc((size_t)N * 4);
    unsigned short* Wlpk = (unsigned short*)alloc((size_t)L * HID * HID * 2);  // packed B layout
    unsigned short* Wrpk = (unsigned short*)alloc((size_t)L * HID * HID * 2);
    unsigned short* Wepk = (unsigned short*)alloc((size_t)L * HID * HID * 2);
    float* mean   = (float*)alloc(HID * 4);
    float* var    = (float*)alloc(HID * 4);
    float* t1     = (float*)alloc((size_t)B * 256 * 4);
    float* t2     = (float*)alloc((size_t)B * 128 * 4);
    float* t3     = (float*)alloc((size_t)B * 64 * 4);
    (void)ws_size; (void)n_in; (void)out_size;

    const int TB = 256;
    auto blocks = [](size_t n, int tb) { return (unsigned)((n + tb - 1) / tb); };

    // 1) edge features + self-loop mean
    k_edge_proj<<<blocks((size_t)E * HID, TB), TB, 0, stream>>>(ea, epW, epB, ee, E);
    k_fill_f32<<<blocks((size_t)N * HID, TB), TB, 0, stream>>>(ee + (size_t)E * HID, (size_t)N * HID, 0.f);
    k_fill_f32<<<blocks((size_t)N, TB), TB, 0, stream>>>(cnt, (size_t)N, 0.f);
    k_selfloop_accum<<<blocks((size_t)E * HID, TB), TB, 0, stream>>>(ee, ei, ee + (size_t)E * HID, cnt, E);
    k_selfloop_div<<<blocks((size_t)N * HID, TB), TB, 0, stream>>>(ee + (size_t)E * HID, cnt, N);
    k_f32_to_bf16<<<blocks((size_t)Et * HID, TB), TB, 0, stream>>>(ee, eebf, (size_t)Et * HID);

    // 2) node projection
    k_node_proj<<<blocks((size_t)N * HID, TB), TB, 0, stream>>>(x, npW, npB, hA, N);

    // 3) weights -> packed bf16 WMMA-B layout (all layers at once)
    k_pack_w<<<blocks((size_t)L * HID * HID, TB), TB, 0, stream>>>(Wl, Wlpk, L);
    k_pack_w<<<blocks((size_t)L * HID * HID, TB), TB, 0, stream>>>(Wr, Wrpk, L);
    k_pack_w<<<blocks((size_t)L * HID * HID, TB), TB, 0, stream>>>(We, Wepk, L);

    float* hcur = hA;
    float* hnxt = hB;
    const int nodeTiles = (N + 15) / 16;
    const int edgeTiles = (Et + 15) / 16;
    const unsigned gemmNodeBlocks = (unsigned)((nodeTiles + MT_PER_BLOCK - 1) / MT_PER_BLOCK);
    const unsigned gemmEdgeBlocks = (unsigned)((edgeTiles + MT_PER_BLOCK - 1) / MT_PER_BLOCK);

    for (int i = 0; i < L; ++i) {
        // node/edge GEMMs via WMMA (bf16 in, f32 accumulate)
        k_f32_to_bf16<<<blocks((size_t)N * HID, TB), TB, 0, stream>>>(hcur, hbf, (size_t)N * HID);
        k_gemm_bf16_k128<<<gemmNodeBlocks, TB, 0, stream>>>(hbf, Wlpk + (size_t)i * HID * HID, bl + i * HID, xl, N);
        k_gemm_bf16_k128<<<gemmNodeBlocks, TB, 0, stream>>>(hbf, Wrpk + (size_t)i * HID * HID, br + i * HID, xr, N);
        k_gemm_bf16_k128<<<gemmEdgeBlocks, TB, 0, stream>>>(eebf, Wepk + (size_t)i * HID * HID, nullptr, em, Et);

        // softmax over incoming edges
        k_fill_u32<<<blocks((size_t)N * HEADS, TB), TB, 0, stream>>>(emax, (size_t)N * HEADS, 0u);
        k_fill_f32<<<blocks((size_t)N * HEADS, TB), TB, 0, stream>>>(denom, (size_t)N * HEADS, 0.f);
        k_fill_f32<<<blocks((size_t)N * HID, TB), TB, 0, stream>>>(gout, (size_t)N * HID, 0.f);
        k_edge_logits<<<blocks((size_t)Et * HEADS, TB), TB, 0, stream>>>(xl, xr, em, ei, att + i * HID, logits, emax, E, N);
        k_edge_accum<<<blocks((size_t)Et * HID, TB), TB, 0, stream>>>(xl, ei, logits, emax, denom, gout, E, N);
        k_gat_finalize<<<blocks((size_t)N * HID, TB), TB, 0, stream>>>(gout, denom, gbias + i * HID, N);

        // BN + relu + residual
        k_bn_stats<<<HID, TB, 0, stream>>>(gout, mean, var, N);
        k_bn_apply_relu_res<<<blocks((size_t)N * HID, TB), TB, 0, stream>>>(gout, mean, var, bng + i * HID, bnb + i * HID, hcur, hnxt, N);
        float* tmp = hcur; hcur = hnxt; hnxt = tmp;
    }

    // 4) head MLP
    k_head_lin1<<<blocks((size_t)B * 256, TB), TB, 0, stream>>>(hcur, mut, pert, r1W, r1b, t1);
    k_head_bn_relu<<<1, 256, 0, stream>>>(t1, bn1g, bn1b, B, 256);
    k_head_lin<<<blocks((size_t)B * 128, TB), TB, 0, stream>>>(t1, r2W, r2b, t2, B, 256, 128, 0);
    k_head_bn_relu<<<1, 128, 0, stream>>>(t2, bn2g, bn2b, B, 128);
    k_head_lin<<<blocks((size_t)B * 64, TB), TB, 0, stream>>>(t2, r3W, r3b, t3, B, 128, 64, 1);
    k_head_out<<<1, 64, 0, stream>>>(t3, clsW, clsb, regW, regb, out);
}
